// AttnCRFDecoder_72430328479852
// MI455X (gfx1250) — compile-verified, matrix operand order checked by reading, not run on hardware
//
#include <hip/hip_runtime.h>
#include <hip/hip_bf16.h>

// ---------------------------------------------------------------------------
// AttnCRFDecoder for MI455X (gfx1250):
//  - bf16 WMMA (v_wmma_f32_16x16x32_bf16) GEMMs + flash attention, fp32 acc
//  - 8-wave workgroups; B-panels / K,V chunks staged to LDS with
//    global_load_async_to_lds_b128 (ASYNCcnt), double buffered
//  - GEMM waves compute 32x64 tiles: 8 WMMA per staged B fragment
//  - CRF forward algorithm in LDS (valid-first sort == skip invalid tokens)
// ---------------------------------------------------------------------------

typedef __bf16 bf16;
typedef bf16  v16bf __attribute__((ext_vector_type(16)));
typedef bf16  v8bf  __attribute__((ext_vector_type(8)));
typedef float v8f   __attribute__((ext_vector_type(8)));

#define WMMA_BF16(A, B, C) \
  __builtin_amdgcn_wmma_f32_16x16x32_bf16(false, (A), false, (B), (short)0, (C), false, false)

#define NB   32
#define NS   512
#define ND   768
#define NH   12
#define HD   64
#define NLAB 9
#define NLC  11        // CRF label space
#define CSTART 9
#define CEND   10

// ---- CDNA5 async global->LDS copy (16B per lane), tracked by ASYNCcnt ----
static __device__ inline void async_cp16(const bf16* g, bf16* l) {
  unsigned int       loff = (unsigned int)(uintptr_t)l;        // addr[31:0] = LDS offset
  unsigned long long ga   = (unsigned long long)(uintptr_t)g;
  asm volatile("global_load_async_to_lds_b128 %0, %1, off" :: "v"(loff), "v"(ga) : "memory");
}
#define ASYNC_WAIT_0() asm volatile("s_wait_asynccnt 0x0" ::: "memory")
#define ASYNC_WAIT_1() asm volatile("s_wait_asynccnt 0x1" ::: "memory")
#define ASYNC_WAIT_2() asm volatile("s_wait_asynccnt 0x2" ::: "memory")

// Build a 16-element bf16 A/B fragment for v_wmma_*_16x16x32 from two
// contiguous 16-byte chunks (CDNA5 ISA 7.12.2: VGPRs 0-3 hold K=8*half+0..7,
// VGPRs 4-7 hold K=16+8*half+0..7, packed 2 per dword).
static __device__ inline v16bf load_frag16(const bf16* p0, const bf16* p1) {
  v8bf lo = *(const v8bf*)p0;
  v8bf hi = *(const v8bf*)p1;
  v16bf r;
#pragma unroll
  for (int i = 0; i < 8; ++i) { r[i] = lo[i]; r[8 + i] = hi[i]; }
  return r;
}

// ---------------- conversion kernels ----------------
__global__ void cvt_x_kernel(const float* __restrict__ X, bf16* __restrict__ Xb, int n) {
  int i = blockIdx.x * 256 + threadIdx.x;
  if (i < n) Xb[i] = (bf16)X[i];
}

// WT  : (2304, 768), WT[c][d] = W{q|k|v}[h][d][kk];  WoT : (768,768), WoT[n][c] = Wo[c][n]
__global__ void cvt_w_kernel(const float* __restrict__ Wq, const float* __restrict__ Wk,
                             const float* __restrict__ Wv, const float* __restrict__ Wo,
                             bf16* __restrict__ WT, bf16* __restrict__ WoT) {
  int i = blockIdx.x * 256 + threadIdx.x;
  const int nWT = 2304 * ND;
  if (i < nWT) {
    int c = i / ND, d = i % ND;
    int qkv = c / ND, rest = c % ND;
    int h = rest / HD, kk = rest % HD;
    const float* W = (qkv == 0) ? Wq : (qkv == 1) ? Wk : Wv;
    WT[i] = (bf16)W[((size_t)h * ND + d) * HD + kk];
  } else {
    int j = i - nWT;
    if (j < ND * ND) {
      int n = j / ND, c = j % ND;
      WoT[j] = (bf16)Wo[(size_t)c * ND + n];
    }
  }
}

// ---------------- fused QKV projection GEMM ----------------
// 256 threads = 8 waves; block computes a 256x64 tile of (BS x 2304); each
// wave owns 32 rows (two A fragments, eight accumulators -> 8 WMMA per
// staged B fragment). B-panel chunk (64 cols x 32 k = 4KB) async-staged to
// LDS, double buffered, shared by all 8 waves.
__global__ void __launch_bounds__(256)
qkv_gemm_kernel(const bf16* __restrict__ Xb, const bf16* __restrict__ WT,
                bf16* __restrict__ Qb, bf16* __restrict__ Kb, bf16* __restrict__ Vt) {
  __shared__ __align__(16) bf16 bs[2][64 * 32];
  const int tid  = threadIdx.x;
  const int lane = tid & 31, w = tid >> 5;
  const int n    = lane & 15;
  const int half = lane >> 4;
  const int r0   = blockIdx.x * 256 + w * 32;  // this wave's 32-row tile
  const int c0   = blockIdx.y * 64;            // output col tile (of 2304)

  const int f   = tid * 16;                    // byte slot in 4KB chunk
  const int col = f >> 6, kb = f & 63;         // B row / byte within 64B row
  auto stage = [&](int k0, int buf) {
    async_cp16(WT + (size_t)(c0 + col) * ND + k0 + (kb >> 1), &bs[buf][col * 32 + (kb >> 1)]);
  };
  stage(0, 0);

  v8f acc[2][4];
#pragma unroll
  for (int m = 0; m < 2; ++m)
#pragma unroll
    for (int t = 0; t < 4; ++t)
#pragma unroll
      for (int r = 0; r < 8; ++r) acc[m][t][r] = 0.f;

  const bf16* arow0 = Xb + (size_t)(r0 + n) * ND;        // A rows 0..15
  const bf16* arow1 = Xb + (size_t)(r0 + 16 + n) * ND;   // A rows 16..31
  const int NIT = ND / 32;                               // 24 k-chunks
  for (int i = 0; i < NIT; ++i) {
    const int k0 = i * 32;
    if (i + 1 < NIT) { stage(k0 + 32, (i + 1) & 1); ASYNC_WAIT_1(); }
    else             { ASYNC_WAIT_0(); }
    __syncthreads();
    v16bf a0 = load_frag16(arow0 + k0 + 8 * half, arow0 + k0 + 16 + 8 * half);
    v16bf a1 = load_frag16(arow1 + k0 + 8 * half, arow1 + k0 + 16 + 8 * half);
    const bf16* Bs = bs[i & 1];
#pragma unroll
    for (int t = 0; t < 4; ++t) {
      const bf16* bcol = Bs + (16 * t + n) * 32;         // B: lane col = output col
      v16bf b = load_frag16(bcol + 8 * half, bcol + 16 + 8 * half);
      acc[0][t] = WMMA_BF16(a0, b, acc[0][t]);
      acc[1][t] = WMMA_BF16(a1, b, acc[1][t]);
    }
    __syncthreads();                                     // protect buffer for next issue
  }

  const int qkv = blockIdx.y / NH;
  const int h   = blockIdx.y % NH;
#pragma unroll
  for (int m = 0; m < 2; ++m) {
#pragma unroll
    for (int t = 0; t < 4; ++t) {
#pragma unroll
      for (int r = 0; r < 8; ++r) {
        int row = r0 + 16 * m + r + 8 * half;            // C layout: M = vgpr + 8*half
        int kk  = 16 * t + n;                            //           N = lane&15
        int b   = row >> 9, s = row & (NS - 1);
        bf16 v = (bf16)acc[m][t][r];
        if (qkv == 0)      Qb[((size_t)(b * NH + h) * NS + s) * HD + kk] = v;
        else if (qkv == 1) Kb[((size_t)(b * NH + h) * NS + s) * HD + kk] = v;
        else               Vt[((size_t)(b * NH + h) * HD + kk) * NS + s] = v;  // transposed
      }
    }
  }
}

// ---------------- flash attention ----------------
// 8 waves x 16 queries = 128 queries per block; K chunk (32x64) and V^T chunk
// (64x32) async-staged to LDS (double buffered), shared across waves.
// Scores computed as K*Q^T so the f32 C-tile (query=lane&15, key=vgpr+8*half)
// feeds the PV WMMA B-operand directly (no cross-lane movement).
__global__ void __launch_bounds__(256)
attn_kernel(const bf16* __restrict__ Qb, const bf16* __restrict__ Kb,
            const bf16* __restrict__ Vt, bf16* __restrict__ Ctx) {
  __shared__ __align__(16) bf16 sm[2][4096];       // [0..2047]=K chunk, [2048..]=V chunk
  const int tid  = threadIdx.x;
  const int lane = tid & 31, w = tid >> 5;
  const int n    = lane & 15;
  const int half = lane >> 4;
  const int s0   = blockIdx.x * 128;
  const int bh   = blockIdx.y;                     // b*NH + h
  const bf16* Q = Qb + (size_t)bh * NS * HD;
  const bf16* K = Kb + (size_t)bh * NS * HD;
  const bf16* V = Vt + (size_t)bh * HD * NS;       // [vd][s]

  const int f  = tid * 16;                         // byte slot in 4KB
  const int vd = f >> 6, sub = (f & 63) >> 1;      // V row / element within 32-elem row
  auto stage = [&](int kc, int buf) {
    async_cp16(K + (size_t)kc * HD + (f >> 1), &sm[buf][f >> 1]);                 // contiguous
    async_cp16(V + (size_t)vd * NS + kc + sub, &sm[buf][2048 + vd * 32 + sub]);   // strided rows
  };
  stage(0, 0);

  // B operand of score GEMM: Q^T for this wave's 16 queries (kd = K dim = 64)
  const bf16* qrow = Q + (size_t)(s0 + w * 16 + n) * HD;
  v16bf bq0 = load_frag16(qrow +      8 * half, qrow + 16 + 8 * half);
  v16bf bq1 = load_frag16(qrow + 32 + 8 * half, qrow + 48 + 8 * half);

  v8f o[4];
#pragma unroll
  for (int t = 0; t < 4; ++t)
#pragma unroll
    for (int r = 0; r < 8; ++r) o[t][r] = 0.f;
  float mrow = -3.0e38f, lrow = 0.f;

  const int NIT = NS / 32;                         // 16 key chunks
  for (int i = 0; i < NIT; ++i) {
    const int kc = i * 32;
    if (i + 1 < NIT) { stage(kc + 32, (i + 1) & 1); ASYNC_WAIT_2(); }
    else             { ASYNC_WAIT_0(); }
    __syncthreads();
    const bf16* Ks = sm[i & 1];
    const bf16* Vs = sm[i & 1] + 2048;

    v8f sc[2];
#pragma unroll
    for (int j = 0; j < 2; ++j) {
      const bf16* krow = Ks + (size_t)(16 * j + n) * HD;   // A: lane row = key
      v16bf a0 = load_frag16(krow +      8 * half, krow + 16 + 8 * half);
      v16bf a1 = load_frag16(krow + 32 + 8 * half, krow + 48 + 8 * half);
      v8f c;
#pragma unroll
      for (int r = 0; r < 8; ++r) c[r] = 0.f;
      c = WMMA_BF16(a0, bq0, c);
      c = WMMA_BF16(a1, bq1, c);
#pragma unroll
      for (int r = 0; r < 8; ++r) sc[j][r] = c[r] * 0.125f;   // 1/sqrt(64)
    }
    // online softmax (query = lane&15; halves combined via shfl_xor 16)
    float cm = -3.0e38f;
#pragma unroll
    for (int j = 0; j < 2; ++j)
#pragma unroll
      for (int r = 0; r < 8; ++r) cm = fmaxf(cm, sc[j][r]);
    cm = fmaxf(cm, __shfl_xor(cm, 16, 32));
    float mnew = fmaxf(mrow, cm);
    float corr = __expf(mrow - mnew);
    float ls = 0.f;
    v8f p[2];
#pragma unroll
    for (int j = 0; j < 2; ++j)
#pragma unroll
      for (int r = 0; r < 8; ++r) {
        float e = __expf(sc[j][r] - mnew);
        p[j][r] = e;
        ls += e;
      }
    ls += __shfl_xor(ls, 16, 32);
    lrow = lrow * corr + ls;
    mrow = mnew;
#pragma unroll
    for (int t = 0; t < 4; ++t)
#pragma unroll
      for (int r = 0; r < 8; ++r) o[t][r] *= corr;
    // pack P^T as WMMA B operand (K=32 keys, N=16 queries)
    v16bf bp;
#pragma unroll
    for (int i2 = 0; i2 < 8; ++i2) { bp[i2] = (bf16)p[0][i2]; bp[8 + i2] = (bf16)p[1][i2]; }
    // O^T += V^T * P^T : A = V^T tiles (M = vd, K = keys) from LDS
#pragma unroll
    for (int t = 0; t < 4; ++t) {
      const bf16* vrow = Vs + (size_t)(16 * t + n) * 32;
      v16bf av = load_frag16(vrow + 8 * half, vrow + 16 + 8 * half);
      o[t] = WMMA_BF16(av, bp, o[t]);
    }
    __syncthreads();                               // protect buffer for next issue
  }

  const float inv = 1.f / lrow;
  const int b = bh / NH, h = bh % NH;
#pragma unroll
  for (int t = 0; t < 4; ++t)
#pragma unroll
    for (int r = 0; r < 8; ++r) {
      int vd2 = 16 * t + r + 8 * half;
      Ctx[((size_t)b * NS + s0 + w * 16 + n) * ND + h * HD + vd2] = (bf16)(o[t][r] * inv);
    }
}

// ---------------- output projection + bias + residual ----------------
// Same 256x64 block / 32x64 wave tiling as qkv_gemm_kernel.
__global__ void __launch_bounds__(256)
out_gemm_kernel(const bf16* __restrict__ Ctx, const bf16* __restrict__ WoT,
                const float* __restrict__ bo, const float* __restrict__ Xf,
                float* __restrict__ Y) {
  __shared__ __align__(16) bf16 bs[2][64 * 32];
  const int tid  = threadIdx.x;
  const int lane = tid & 31, w = tid >> 5;
  const int n    = lane & 15;
  const int half = lane >> 4;
  const int r0   = blockIdx.x * 256 + w * 32;
  const int c0   = blockIdx.y * 64;

  const int f   = tid * 16;
  const int col = f >> 6, kb = f & 63;
  auto stage = [&](int k0, int buf) {
    async_cp16(WoT + (size_t)(c0 + col) * ND + k0 + (kb >> 1), &bs[buf][col * 32 + (kb >> 1)]);
  };
  stage(0, 0);

  v8f acc[2][4];
#pragma unroll
  for (int m = 0; m < 2; ++m)
#pragma unroll
    for (int t = 0; t < 4; ++t)
#pragma unroll
      for (int r = 0; r < 8; ++r) acc[m][t][r] = 0.f;

  const bf16* arow0 = Ctx + (size_t)(r0 + n) * ND;
  const bf16* arow1 = Ctx + (size_t)(r0 + 16 + n) * ND;
  const int NIT = ND / 32;
  for (int i = 0; i < NIT; ++i) {
    const int k0 = i * 32;
    if (i + 1 < NIT) { stage(k0 + 32, (i + 1) & 1); ASYNC_WAIT_1(); }
    else             { ASYNC_WAIT_0(); }
    __syncthreads();
    v16bf a0 = load_frag16(arow0 + k0 + 8 * half, arow0 + k0 + 16 + 8 * half);
    v16bf a1 = load_frag16(arow1 + k0 + 8 * half, arow1 + k0 + 16 + 8 * half);
    const bf16* Bs = bs[i & 1];
#pragma unroll
    for (int t = 0; t < 4; ++t) {
      const bf16* bcol = Bs + (16 * t + n) * 32;
      v16bf b = load_frag16(bcol + 8 * half, bcol + 16 + 8 * half);
      acc[0][t] = WMMA_BF16(a0, b, acc[0][t]);
      acc[1][t] = WMMA_BF16(a1, b, acc[1][t]);
    }
    __syncthreads();
  }
#pragma unroll
  for (int m = 0; m < 2; ++m)
#pragma unroll
    for (int t = 0; t < 4; ++t)
#pragma unroll
      for (int r = 0; r < 8; ++r) {
        int row = r0 + 16 * m + r + 8 * half;
        int c   = c0 + 16 * t + n;
        Y[(size_t)row * ND + c] = acc[m][t][r] + bo[c] + Xf[(size_t)row * ND + c];
      }
}

// ---------------- LayerNorm + emission logits ----------------
__global__ void ln_logits_kernel(const float* __restrict__ Y, const float* __restrict__ g,
                                 const float* __restrict__ bta, const float* __restrict__ Wl,
                                 const float* __restrict__ bl, float* __restrict__ Lg) {
  __shared__ float xn[ND];
  __shared__ float red[256];
  const int row = blockIdx.x, t = threadIdx.x;
  const float* y = Y + (size_t)row * ND;

  float s = 0.f, s2 = 0.f;
  for (int i = t; i < ND; i += 256) { float v = y[i]; s += v; s2 += v * v; }
  red[t] = s; __syncthreads();
  for (int o = 128; o > 0; o >>= 1) { if (t < o) red[t] += red[t + o]; __syncthreads(); }
  const float mu = red[0] / ND; __syncthreads();
  red[t] = s2; __syncthreads();
  for (int o = 128; o > 0; o >>= 1) { if (t < o) red[t] += red[t + o]; __syncthreads(); }
  const float var  = red[0] / ND - mu * mu;
  const float rstd = rsqrtf(var + 1e-5f);
  for (int i = t; i < ND; i += 256) xn[i] = (y[i] - mu) * rstd * g[i] + bta[i];
  __syncthreads();

  if (t < NLAB) {
    float d = bl[t];
    for (int i = 0; i < ND; ++i) d += xn[i] * Wl[(size_t)i * NLAB + t];
    Lg[(size_t)row * NLC + t] = d;
  } else if (t < NLC) {
    Lg[(size_t)row * NLC + t] = -1000.f;   // PAD_NEG start/end columns
  }
}

// ---------------- CRF forward + gold score (one block per batch) ----------
// Valid-first stable sort + length mask == process valid tokens in order.
__global__ void crf_kernel(const float* __restrict__ Lg, const int* __restrict__ mask,
                           const int* __restrict__ labels, const float* __restrict__ trans,
                           float* __restrict__ scoreOut) {
  __shared__ float T[NLC * NLC];
  __shared__ float alpha[NLC];
  __shared__ float anew[NLC];
  const int b = blockIdx.x, t = threadIdx.x;
  for (int i = t; i < NLC * NLC; i += 32) T[i] = trans[i];
  if (t < NLC) alpha[t] = (t == CSTART) ? 0.f : -10000.f;
  __syncthreads();

  float gold = 0.f;
  int prevLab = CSTART;
  for (int s = 0; s < NS; ++s) {
    int m = mask[b * NS + s];               // uniform across block
    if (m) {
      const float* lg = Lg + ((size_t)b * NS + s) * NLC;
      if (t < NLC) {
        float mx = -3.0e38f;
        for (int p = 0; p < NLC; ++p) mx = fmaxf(mx, alpha[p] + T[t * NLC + p]);
        float sm = 0.f;
        for (int p = 0; p < NLC; ++p) sm += __expf(alpha[p] + T[t * NLC + p] - mx);
        anew[t] = lg[t] + mx + __logf(sm);
      }
      __syncthreads();
      if (t < NLC) alpha[t] = anew[t];
      if (t == 0) {
        int lab = labels[b * NS + s];
        gold += lg[lab] + T[lab * NLC + prevLab];
        prevLab = lab;
      }
      __syncthreads();
    }
  }
  if (t == 0) {
    gold += T[CEND * NLC + prevLab];
    float mx = -3.0e38f;
    for (int j = 0; j < NLC; ++j) mx = fmaxf(mx, alpha[j] + T[CEND * NLC + j]);
    float sm = 0.f;
    for (int j = 0; j < NLC; ++j) sm += __expf(alpha[j] + T[CEND * NLC + j] - mx);
    scoreOut[b] = gold - (mx + __logf(sm));
  }
}

__global__ void finalize_kernel(const float* __restrict__ scoreOut, float* __restrict__ out) {
  float v = scoreOut[threadIdx.x];          // 32 batches == 32 lanes
#pragma unroll
  for (int o = 16; o > 0; o >>= 1) v += __shfl_xor(v, o, 32);
  if (threadIdx.x == 0) out[0] = -(v / (float)NB);
}

// ---------------------------------------------------------------------------
extern "C" void kernel_launch(void* const* d_in, const int* in_sizes, int n_in,
                              void* d_out, int out_size, void* d_ws, size_t ws_size,
                              hipStream_t stream) {
  const float* X     = (const float*)d_in[0];
  const int*   pmask = (const int*)d_in[1];
  const int*   labs  = (const int*)d_in[2];
  const float* Wq    = (const float*)d_in[3];
  const float* Wk    = (const float*)d_in[4];
  const float* Wv    = (const float*)d_in[5];
  const float* Wo    = (const float*)d_in[6];
  const float* bo    = (const float*)d_in[7];
  const float* ln_g  = (const float*)d_in[8];
  const float* ln_b  = (const float*)d_in[9];
  const float* Wl    = (const float*)d_in[10];
  const float* bl    = (const float*)d_in[11];
  const float* trans = (const float*)d_in[12];
  float* out = (float*)d_out;

  char* ws = (char*)d_ws;
  size_t off = 0;
  auto alloc = [&](size_t bytes) -> void* {
    void* p = ws + off;
    off = (off + bytes + 255) & ~(size_t)255;
    return p;
  };
  const size_t BS = (size_t)NB * NS;
  bf16*  Xb  = (bf16*)alloc(BS * ND * 2);
  bf16*  WT  = (bf16*)alloc((size_t)3 * NH * HD * ND * 2);
  bf16*  WoT = (bf16*)alloc((size_t)ND * ND * 2);
  bf16*  Qb  = (bf16*)alloc((size_t)NB * NH * NS * HD * 2);
  bf16*  Kb  = (bf16*)alloc((size_t)NB * NH * NS * HD * 2);
  bf16*  Vt  = (bf16*)alloc((size_t)NB * NH * HD * NS * 2);
  bf16*  Ctx = (bf16*)alloc(BS * ND * 2);
  float* Y   = (float*)alloc(BS * ND * 4);
  float* Lg  = (float*)alloc(BS * NLC * 4);
  float* Sb  = (float*)alloc(NB * 4);
  (void)ws_size; (void)n_in; (void)in_sizes; (void)out_size;

  {
    int n = (int)(BS * ND);
    cvt_x_kernel<<<(n + 255) / 256, 256, 0, stream>>>(X, Xb, n);
  }
  {
    int n = 3 * NH * HD * ND + ND * ND;
    cvt_w_kernel<<<(n + 255) / 256, 256, 0, stream>>>(Wq, Wk, Wv, Wo, WT, WoT);
  }
  qkv_gemm_kernel<<<dim3(BS / 256, 3 * NH), 256, 0, stream>>>(Xb, WT, Qb, Kb, Vt);
  attn_kernel<<<dim3(NS / 128, NB * NH), 256, 0, stream>>>(Qb, Kb, Vt, Ctx);
  out_gemm_kernel<<<dim3(BS / 256, ND / 64), 256, 0, stream>>>(Ctx, WoT, bo, X, Y);
  ln_logits_kernel<<<(int)BS, 256, 0, stream>>>(Y, ln_g, ln_b, Wl, bl, Lg);
  crf_kernel<<<NB, 32, 0, stream>>>(Lg, pmask, labs, trans, Sb);
  finalize_kernel<<<1, 32, 0, stream>>>(Sb, out);
}
